// EncoderLSTM_25718264168631
// MI455X (gfx1250) — matrix operand order repaired and checked by reading, not compiled, not run on hardware
//
#include <hip/hip_runtime.h>

// ---------------------------------------------------------------------------
// LSTM (B=64, T=512, F=512, H=1024) for gfx1250 (MI455X), wave32 + WMMA bf16.
//   prep:   x -> bf16 [T][B][F];  h0 -> bf16 ping buffer;
//           Wi/Wh -> bf16 packed into per-(WG,ktile,ntile) WMMA B-fragment
//           layout (lane-major, 16 contiguous halves per lane).
//   main:   32 persistent workgroups, each owns 32 hidden units (all 4 gates).
//           Per step: z = [x_t|h] @ Wpack via v_wmma_f32_16x16x32_bf16.
//           A-operand staged to LDS with GLOBAL_LOAD_ASYNC_TO_LDS_B128,
//           double-buffered so staging overlaps WMMA (ASYNCcnt pipeline).
//           Gates in fp32, c in LDS, h -> bf16 ping-pong + fp32 output.
//           Device-wide barrier (atomic counter in workspace) between steps.
// ---------------------------------------------------------------------------

typedef __attribute__((ext_vector_type(16))) __bf16 v16bf;
typedef __attribute__((ext_vector_type(8)))  float  v8f;
typedef __attribute__((ext_vector_type(4)))  int    v4i;

#define B_      64
#define T_      512
#define F_      512
#define H_      1024
#define G4      4096      // 4*H
#define NWG     32
#define SLICE   32        // H / NWG hidden units per workgroup
#define COLS    128       // 4 * SLICE gate columns per workgroup
#define KTOT    1536      // F + H
#define CK      128       // K-chunk staged in LDS
#define NCHUNK  12        // KTOT / CK
#define KT_PER  4         // 32-wide WMMA k-tiles per chunk
#define NKT     48        // KTOT / 32
#define APITCH  136       // CK + 8 halves (LDS bank-conflict padding)

union FragBF { v16bf v; int4 q[2]; };

typedef __attribute__((address_space(1))) v4i gv4i;   // global int4
typedef __attribute__((address_space(3))) v4i lv4i;   // LDS int4

// Async global->LDS copy of 16 bytes (per lane), tracked by ASYNCcnt.
__device__ __forceinline__ void async_copy_b128(const void* g, void* l) {
#if __has_builtin(__builtin_amdgcn_global_load_async_to_lds_b128)
  __builtin_amdgcn_global_load_async_to_lds_b128(
      (gv4i*)(unsigned long long)g,
      (lv4i*)(unsigned)(unsigned long long)l, 0, 0);
#else
  unsigned           ldsoff = (unsigned)(unsigned long long)l;
  unsigned long long ga     = (unsigned long long)g;
  asm volatile("global_load_async_to_lds_b128 %0, %1, off"
               :: "v"(ldsoff), "v"(ga) : "memory");
#endif
}

__device__ __forceinline__ void wait_async_le4() {
#if __has_builtin(__builtin_amdgcn_s_wait_asynccnt)
  __builtin_amdgcn_s_wait_asynccnt(4);
#else
  asm volatile("s_wait_asynccnt 0x4" ::: "memory");
#endif
}
__device__ __forceinline__ void wait_async_le0() {
#if __has_builtin(__builtin_amdgcn_s_wait_asynccnt)
  __builtin_amdgcn_s_wait_asynccnt(0);
#else
  asm volatile("s_wait_asynccnt 0x0" ::: "memory");
#endif
}

__device__ __forceinline__ unsigned short f2bf(float f) {
  unsigned u = __float_as_uint(f);
  u += 0x7FFFu + ((u >> 16) & 1u);   // round-to-nearest-even
  return (unsigned short)(u >> 16);
}
__device__ __forceinline__ float sigmoidf_(float x) {
  return 1.0f / (1.0f + __expf(-x));
}
__device__ __forceinline__ float tanhf_(float x) {
  float e = __expf(-2.0f * x);
  return (1.0f - e) / (1.0f + e);
}

// ------------------------------- prep kernels ------------------------------

__global__ void k_init(unsigned* ctr) {
  if (threadIdx.x == 0) *ctr = 0u;
}

// x: [B][T][F] fp32 -> xbf: [T][B][F] bf16 (t-major so each step is a block)
__global__ void k_conv_x(const float* __restrict__ x,
                         unsigned short* __restrict__ xbf) {
  const int n = B_ * T_ * F_;
  for (int i = blockIdx.x * blockDim.x + threadIdx.x; i < n;
       i += gridDim.x * blockDim.x) {
    int f = i % F_;
    int b = (i / F_) % B_;
    int t = i / (F_ * B_);
    xbf[i] = f2bf(x[((size_t)b * T_ + t) * F_ + f]);
  }
}

__global__ void k_conv_h0(const float* __restrict__ h0,
                          unsigned short* __restrict__ hbuf0) {
  int i = blockIdx.x * blockDim.x + threadIdx.x;
  if (i < B_ * H_) hbuf0[i] = f2bf(h0[i]);
}

// Pack [Wi;Wh] (KTOT x 4H fp32) into bf16 WMMA-B-fragment-ready layout:
//   wp[wg][ktg][nt][lane][e]  (e = 16 contiguous halves per lane)
//   lane<16: n=lane,    k_local = e<8 ? e : 8+e      (K 0..7, 16..23)
//   lane>=16: n=lane-16, k_local = e<8 ? 8+e : 16+e  (K 8..15, 24..31)
// Column order per WG is gate-major: [i(32) | f(32) | g(32) | o(32)] over the
// WG's hidden slice, so the elementwise update is WG-local.
__global__ void k_pack_w(const float* __restrict__ Wi,
                         const float* __restrict__ Wh,
                         unsigned short* __restrict__ wp) {
  const int n = NWG * NKT * 8 * 32 * 16;
  for (int i = blockIdx.x * blockDim.x + threadIdx.x; i < n;
       i += gridDim.x * blockDim.x) {
    int e    = i & 15;
    int lane = (i >> 4) & 31;
    int nt   = (i >> 9) & 7;
    int ktg  = (i >> 12) % NKT;
    int wg   = (i >> 12) / NKT;
    int kl   = ((lane >> 4) << 3) + (e & 7) + ((e >> 3) << 4);  // 0..31
    int k    = ktg * 32 + kl;                    // 0..1535
    int col  = nt * 16 + (lane & 15);            // 0..127
    int gate = col >> 5;
    int jj   = col & 31;
    int G    = gate * H_ + wg * SLICE + jj;      // column in 4H
    float v = (k < F_) ? Wi[(size_t)k * G4 + G]
                       : Wh[(size_t)(k - F_) * G4 + G];
    wp[i] = f2bf(v);
  }
}

// ------------------------------- main kernel -------------------------------

__global__ void __launch_bounds__(256, 1)
k_lstm(const float* __restrict__ c0,
       const float* __restrict__ bh,
       const unsigned short* __restrict__ xbf,
       const unsigned short* __restrict__ wp,
       unsigned short* __restrict__ hping,   // 2 x [B][H] bf16, ping-pong
       unsigned* __restrict__ ctr,
       float* __restrict__ out) {            // [B][T][H] fp32
  __shared__ union {
    unsigned short a[2][B_ * APITCH];        // double-buffered A chunks (bf16)
    float          z[B_ * COLS];             // gate pre-activations (fp32)
  } sm;
  __shared__ float c_lds[B_ * SLICE];        // persistent cell state

  const int tid  = threadIdx.x;
  const int wg   = blockIdx.x;
  const int w    = tid >> 5;                 // wave id 0..7 -> N-tile
  const int lane = tid & 31;
  const int j0   = wg * SLICE;

  // init cell state from c0
  for (int i = tid; i < B_ * SLICE; i += 256) {
    int b = i >> 5, jj = i & 31;
    c_lds[i] = c0[(size_t)b * H_ + j0 + jj];
  }

  // A-fragment per-lane geometry (16x32 bf16, ISA layout):
  //  lanes 0-15: M=lane,    K in {0..7, 16..23}
  //  lanes16-31: M=lane-16, K in {8..15, 24..31}
  const int arow = lane & 15;
  const int ak   = (lane >> 4) * 8;

  // this wave's packed-weight base: tile = [wg][ktg][nt=w], 512 halves/tile
  const unsigned short* wpw =
      wp + ((size_t)(wg * NKT) * 8 + w) * 512 + (size_t)lane * 16;

  for (int t = 0; t < T_; ++t) {
    const unsigned short* hprev = hping + (size_t)(t & 1) * (B_ * H_);
    unsigned short*       hnext = hping + (size_t)((t + 1) & 1) * (B_ * H_);
    const unsigned short* xt    = xbf + (size_t)t * (B_ * F_);

    v8f acc[4];
#pragma unroll
    for (int m = 0; m < 4; ++m)
#pragma unroll
      for (int v = 0; v < 8; ++v) acc[m][v] = 0.0f;

    // --- async staging of one A chunk (64 rows x 128 halves), 4 b128/lane --
    auto stage_chunk = [&](int ch) {
      const int k0 = ch * CK;
      const unsigned short* src;
      int stride;
      if (k0 < F_) { src = xt + k0;           stride = F_; }
      else         { src = hprev + (k0 - F_); stride = H_; }
      unsigned short* dst = sm.a[ch & 1];
#pragma unroll
      for (int i = 0; i < 4; ++i) {
        int u  = tid + i * 256;              // 1024 x 16B units
        int r  = u >> 3;
        int c8 = (u & 7) * 8;                // halves
        async_copy_b128(src + (size_t)r * stride + c8,
                        dst + r * APITCH + c8);
      }
    };

    // prologue: chunks 0 and 1 in flight (8 async instrs per wave)
    stage_chunk(0);
    stage_chunk(1);

    for (int ch = 0; ch < NCHUNK; ++ch) {
      // retire chunk ch (async loads complete in order; 4 instrs per chunk)
      if (ch + 1 < NCHUNK) wait_async_le4();
      else                 wait_async_le0();
      __syncthreads();                       // chunk ch visible to all waves

      const unsigned short* abuf = sm.a[ch & 1];
#pragma unroll
      for (int kt = 0; kt < KT_PER; ++kt) {
        const int ktg = ch * KT_PER + kt;
        FragBF bf;
        const int4* bp = (const int4*)(wpw + (size_t)ktg * (8 * 512));
        bf.q[0] = bp[0];
        bf.q[1] = bp[1];
#pragma unroll
        for (int mt = 0; mt < 4; ++mt) {
          FragBF af;
          const unsigned short* ap =
              &abuf[(mt * 16 + arow) * APITCH + kt * 32 + ak];
          af.q[0] = *(const int4*)ap;
          af.q[1] = *(const int4*)(ap + 16);
          acc[mt] = __builtin_amdgcn_wmma_f32_16x16x32_bf16(
              false, af.v, false, bf.v, (short)0, acc[mt], false, false);
        }
      }
      __syncthreads();                       // all waves done reading buffer
      if (ch + 2 < NCHUNK) stage_chunk(ch + 2);  // refill freed buffer
    }

    // scatter accumulators to z LDS (D layout: lane->col, vgpr->row)
    {
      const int colb = w * 16 + (lane & 15);
      const int rowb = (lane >> 4) * 8;
#pragma unroll
      for (int mt = 0; mt < 4; ++mt)
#pragma unroll
        for (int v = 0; v < 8; ++v)
          sm.z[(mt * 16 + rowb + v) * COLS + colb] = acc[mt][v];
    }
    __syncthreads();

    // gates + state update for this WG's 64x32 (batch x hidden) slice
    for (int i = tid; i < B_ * SLICE; i += 256) {
      int b = i >> 5, jj = i & 31;
      float iv = sm.z[b * COLS + jj]      + bh[0 * H_ + j0 + jj];
      float fv = sm.z[b * COLS + 32 + jj] + bh[1 * H_ + j0 + jj];
      float gv = sm.z[b * COLS + 64 + jj] + bh[2 * H_ + j0 + jj];
      float ov = sm.z[b * COLS + 96 + jj] + bh[3 * H_ + j0 + jj];
      float ig = sigmoidf_(iv);
      float fg = sigmoidf_(fv);
      float gg = tanhf_(gv);
      float og = sigmoidf_(ov);
      float cn = fg * c_lds[i] + ig * gg;
      c_lds[i] = cn;
      float hn = og * tanhf_(cn);
      hnext[(size_t)b * H_ + j0 + jj] = f2bf(hn);
      out[((size_t)b * T_ + t) * H_ + j0 + jj] = hn;
    }

    // device-wide step barrier (monotonic counter, zeroed per launch)
    __threadfence();
    __syncthreads();
    if (tid == 0) {
      atomicAdd(ctr, 1u);
      const unsigned tgt = (unsigned)NWG * (unsigned)(t + 1);
      while (__hip_atomic_load(ctr, __ATOMIC_RELAXED,
                               __HIP_MEMORY_SCOPE_AGENT) < tgt) {
        __builtin_amdgcn_s_sleep(2);
      }
    }
    __syncthreads();
    __threadfence();
  }
}

// ------------------------------- launcher ----------------------------------

extern "C" void kernel_launch(void* const* d_in, const int* in_sizes, int n_in,
                              void* d_out, int out_size, void* d_ws,
                              size_t ws_size, hipStream_t stream) {
  (void)in_sizes; (void)n_in; (void)out_size; (void)ws_size;
  const float* x  = (const float*)d_in[0];
  const float* c0 = (const float*)d_in[1];
  const float* h0 = (const float*)d_in[2];
  const float* Wi = (const float*)d_in[3];
  const float* Wh = (const float*)d_in[4];
  const float* bh = (const float*)d_in[5];
  float* out = (float*)d_out;

  char* ws = (char*)d_ws;
  const size_t off_ctr = 0;
  const size_t off_h   = 256;                                   // barrier ctr
  const size_t off_x   = off_h + (size_t)2 * B_ * H_ * 2;       // h ping-pong
  const size_t off_w   = off_x + (size_t)B_ * T_ * F_ * 2;      // x bf16
  // off_w + NWG*NKT*8*32*16*2 = ~46.4 MB total workspace

  unsigned*       ctr   = (unsigned*)(ws + off_ctr);
  unsigned short* hping = (unsigned short*)(ws + off_h);
  unsigned short* xbf   = (unsigned short*)(ws + off_x);
  unsigned short* wpk   = (unsigned short*)(ws + off_w);

  k_init<<<1, 32, 0, stream>>>(ctr);
  k_conv_h0<<<(B_ * H_ + 255) / 256, 256, 0, stream>>>(h0, hping);
  k_conv_x<<<4096, 256, 0, stream>>>(x, xbf);
  k_pack_w<<<4096, 256, 0, stream>>>(Wi, Wh, wpk);
  k_lstm<<<NWG, 256, 0, stream>>>(c0, bh, xbf, wpk, hping, ctr, out);
}
